// HMoENetwork_86955907875107
// MI455X (gfx1250) — matrix operand order repaired and checked by reading, not compiled, not run on hardware
//
#include <hip/hip_runtime.h>
#include <hip/hip_bf16.h>
#include <math.h>

// ---------------------------------------------------------------------------
// Problem constants
// ---------------------------------------------------------------------------
#define BB   8192
#define IN_  1024
#define DD   8
#define SS   4
#define EE   2
#define H1_  512
#define H2_  256
#define GG   64
#define NE_  20          // S + D*E
#define GCOLS 576        // 8*64 domain-gate cols + 64 local-gate cols

typedef __attribute__((ext_vector_type(16))) __bf16 v16bf;
typedef __attribute__((ext_vector_type(8)))  float  v8f;
typedef __attribute__((ext_vector_type(4)))  unsigned int v4u;
typedef __attribute__((ext_vector_type(8)))  int    v8i;
typedef __attribute__((ext_vector_type(4)))  int    v4i;

union Frag16 {
    v16bf v;
    unsigned short s[16];
    uint4 q[2];
};

__device__ __forceinline__ unsigned short f2bf(float f) {
    union { float f; unsigned u; } x; x.f = f;
    unsigned u = x.u;
    unsigned r = u + 0x7fffu + ((u >> 16) & 1u);   // round-to-nearest-even
    return (unsigned short)(r >> 16);
}

__device__ __forceinline__ v8f wmma_bf16(v16bf a, v16bf b, v8f c) {
    // (neg_a, A, neg_b, B, c_mod, C, reuse_a, reuse_b)
    return __builtin_amdgcn_wmma_f32_16x16x32_bf16(false, a, false, b,
                                                   (short)0, c, false, false);
}

// ---------------------------------------------------------------------------
// Stage a contiguous 128 KB block (packed x tiles for one 64-row block) into
// LDS via the Tensor Data Mover. 1-D tensor of 16384 x 8-byte elements.
// Descriptor field layout per CDNA5 ISA 8.3/8.4 (D# group 0 / group 1).
// Issued by wave 0 only; caller must __syncthreads() afterwards.
// NOTE: this toolchain (clang-23 / therock-10.0) exposes the 6-argument
// builtin: (v4u g0, v8i g1, v4i g2, v4i g3, v8i extra, i32 cpol).
// ---------------------------------------------------------------------------
__device__ __forceinline__ void tdm_stage_128k(void* lds_dst, const void* gsrc) {
#if __has_builtin(__builtin_amdgcn_tensor_load_to_lds)
    unsigned lds_off = (unsigned)(unsigned long long)(size_t)lds_dst;
    unsigned long long ga = (unsigned long long)(size_t)gsrc;
    v4u g0;
    g0[0] = 1u;                                            // count=1, user mode
    g0[1] = lds_off;                                       // lds_addr[31:0]
    g0[2] = (unsigned)(ga & 0xFFFFFFFFull);                // global_addr[31:0]
    g0[3] = (unsigned)((ga >> 32) & 0x01FFFFFFull)         // global_addr[56:32]
          | (2u << 30);                                    // type = 2 (image)
    v8i g1;
    g1[0] = (int)(3u << 16);        // workgroup_mask=0, data_size=3 (8 bytes)
    g1[1] = (int)(16384u << 16);    // tensor_dim0[15:0] in bits [63:48]
    g1[2] = (int)(1u << 16);        // tensor_dim0[31:16]=0, tensor_dim1=1
    g1[3] = (int)(16384u << 16);    // tensor_dim1 hi=0, tile_dim0=16384
    g1[4] = 0;                      // tile_dim1=0 (unused), tile_dim2=0
    g1[5] = 16384;                  // tensor_dim0_stride[31:0]
    g1[6] = 0;                      // stride hi, tensor_dim1_stride lo
    g1[7] = 0;                      // tensor_dim1_stride hi
    v4i z4 = {0, 0, 0, 0};          // groups 2/3 unused (<=2D tensor)
    v8i z8 = {0, 0, 0, 0, 0, 0, 0, 0};
    __builtin_amdgcn_tensor_load_to_lds(g0, g1, z4, z4, z8, 0);
    __builtin_amdgcn_s_wait_tensorcnt((short)0);
#else
    // Fallback: plain VGPR round-trip copy by the first wave.
    const uint4* s = (const uint4*)gsrc;
    uint4* d = (uint4*)lds_dst;
    int lane = threadIdx.x & 31;
    for (int i = lane; i < 8192; i += 32) d[i] = s[i];
#endif
}

// ---------------------------------------------------------------------------
// Fragment-linear packing.
// A tile (16x32 bf16): lane L (mrow=L&15) element j -> K = ((j&8)<<1)+(j&7)+((L>>4)<<3)
// B tile (32x16 bf16): lane L (ncol=L&15) element j -> K = ((j&8)<<1)+(j&7)+((L>>4)<<3)
// C tile (16x16 f32) : lane L (N=L&15), reg r -> M = r + ((L>>4)<<3)
// Tiles stored: 1024 B each, per-lane 32 B contiguous at lane*32.
// ---------------------------------------------------------------------------

// Pack x [B x IN] f32 row-major -> A-fragment-linear bf16 (tile idx = mt*32+kt)
__global__ __launch_bounds__(256) void pack_x_kernel(const float* __restrict__ src,
                                                     unsigned short* __restrict__ dst) {
    int lane = threadIdx.x & 31, wave = threadIdx.x >> 5;
    int tile = blockIdx.x * 8 + wave;                 // 16384 tiles total
    if (tile >= (BB / 16) * (IN_ / 32)) return;
    int mt = tile >> 5, kt = tile & 31;
    int mrow = lane & 15, lh = lane >> 4;
    Frag16 t;
#pragma unroll
    for (int j = 0; j < 16; ++j) {
        int k = kt * 32 + ((j & 8) << 1) + (j & 7) + (lh << 3);
        t.s[j] = f2bf(src[(size_t)(mt * 16 + mrow) * IN_ + k]);
    }
    uint4* dp = (uint4*)(dst + (size_t)tile * 512 + (size_t)lane * 16);
    dp[0] = t.q[0];
    dp[1] = t.q[1];
}

// Pack B-operand weight matrices [batch][K x N] f32 row-major ->
// B-fragment-linear bf16, tile idx within batch = nt*nTilesK + kt.
__global__ __launch_bounds__(256) void pack_B_kernel(const float* __restrict__ src,
                                                     unsigned short* __restrict__ dst,
                                                     int N, int nTilesK,
                                                     long srcBatchStride,
                                                     int tilesPerBatch, int totalTiles) {
    int lane = threadIdx.x & 31, wave = threadIdx.x >> 5;
    int tile = blockIdx.x * 8 + wave;
    if (tile >= totalTiles) return;
    int batch = tile / tilesPerBatch;
    int r = tile - batch * tilesPerBatch;
    int nt = r / nTilesK, kt = r - nt * nTilesK;
    int ncol = lane & 15, lh = lane >> 4;
    Frag16 t;
#pragma unroll
    for (int j = 0; j < 16; ++j) {
        int k = kt * 32 + ((j & 8) << 1) + (j & 7) + (lh << 3);
        t.s[j] = f2bf(src[(size_t)batch * srcBatchStride + (size_t)k * N + nt * 16 + ncol]);
    }
    uint4* dp = (uint4*)(dst + (size_t)tile * 512 + (size_t)lane * 16);
    dp[0] = t.q[0];
    dp[1] = t.q[1];
}

// ---------------------------------------------------------------------------
// Gate GEMM: gh[b, 0..575] = relu(x @ [gW1(8 blocks)|lW1] + bias)
// One workgroup = 64 rows (4 m-tiles), 8 waves strided over 36 n-tiles.
// x staged to LDS with GLOBAL_LOAD_ASYNC_TO_LDS_B128 (ASYNCcnt path).
// ---------------------------------------------------------------------------
__global__ __launch_bounds__(256) void gate_gemm_kernel(const unsigned short* __restrict__ xp,
                                                        const unsigned short* __restrict__ gwp,
                                                        const float* __restrict__ gb1,
                                                        const float* __restrict__ lb1,
                                                        float* __restrict__ ghws) {
    extern __shared__ char smem[];
    unsigned short* sx = (unsigned short*)smem;       // 64x1024 bf16 fragment tiles (128 KB)
    int tid = threadIdx.x, lane = tid & 31, wave = tid >> 5;
    int mb = blockIdx.x;
    int l15 = lane & 15, lh = lane >> 4;

    // Stage this row-block's A tiles (contiguous 128 KB) asynchronously to LDS.
    {
        unsigned long long gbase = (unsigned long long)(size_t)(xp + (size_t)mb * 65536);
        unsigned lbase = (unsigned)(unsigned long long)(size_t)sx;
        for (int i = tid; i < 8192; i += 256) {
            unsigned long long ga = gbase + (unsigned long long)i * 16ull;
            unsigned la = lbase + (unsigned)i * 16u;
            asm volatile("global_load_async_to_lds_b128 %0, %1, off"
                         :: "v"(la), "v"(ga) : "memory");
        }
        asm volatile("s_wait_asynccnt 0x0" ::: "memory");
    }
    __syncthreads();

    for (int nt = wave; nt < 36; nt += 8) {
        v8f acc[4];
#pragma unroll
        for (int m = 0; m < 4; ++m) acc[m] = (v8f){0,0,0,0,0,0,0,0};
        int blk = nt >> 2, ntin = nt & 3;
        const unsigned short* wb = gwp + (size_t)(blk * 128 + ntin * 32) * 512;
        for (int kt = 0; kt < 32; ++kt) {
            v16bf bf = *(const v16bf*)(wb + (size_t)kt * 512 + lane * 16);
#pragma unroll
            for (int m = 0; m < 4; ++m) {
                v16bf af = *(const v16bf*)(sx + (size_t)(m * 32 + kt) * 512 + lane * 16);
                acc[m] = wmma_bf16(af, bf, acc[m]);
            }
        }
        int col = ntin * 16 + l15;                         // col within 64-wide block
        float bias = (blk < 8) ? gb1[blk * 64 + col] : lb1[col];
        int ng = nt * 16 + l15;                            // global gate column 0..575
#pragma unroll
        for (int m = 0; m < 4; ++m) {
#pragma unroll
            for (int r = 0; r < 8; ++r) {
                int b = mb * 64 + m * 16 + r + 8 * lh;
                ghws[(size_t)b * GCOLS + ng] = fmaxf(acc[m][r] + bias, 0.f);
            }
        }
    }
}

// ---------------------------------------------------------------------------
// Gate finish: per row b, softmax(gh_d @ gW2_d + gb2_d) (6-way) for each domain,
// lg = softmax(lh @ lW2 + lb2) (8-way), then combine to w[b, 0..19].
// One wave per row; butterfly reductions over 32 lanes.
// ---------------------------------------------------------------------------
__global__ __launch_bounds__(256) void gate_finish_kernel(const float* __restrict__ ghws,
                                                          const float* __restrict__ gW2,
                                                          const float* __restrict__ gb2,
                                                          const float* __restrict__ lW2,
                                                          const float* __restrict__ lb2,
                                                          float* __restrict__ wws) {
    __shared__ float sg[8][48];   // per-wave domain gates [d*6+o]
    __shared__ float slg[8][8];   // per-wave local gate
    int lane = threadIdx.x & 31, wave = threadIdx.x >> 5;
    int b = blockIdx.x * 8 + wave;
    const float* row = ghws + (size_t)b * GCOLS;

    // domain gates
    for (int d = 0; d < 8; ++d) {
        float a0 = row[d * 64 + lane];
        float a1 = row[d * 64 + 32 + lane];
        float s[6];
#pragma unroll
        for (int o = 0; o < 6; ++o)
            s[o] = a0 * gW2[(d * 64 + lane) * 6 + o] + a1 * gW2[(d * 64 + 32 + lane) * 6 + o];
#pragma unroll
        for (int m = 16; m >= 1; m >>= 1)
#pragma unroll
            for (int o = 0; o < 6; ++o) s[o] += __shfl_xor(s[o], m, 32);
#pragma unroll
        for (int o = 0; o < 6; ++o) s[o] += gb2[d * 6 + o];
        float mx = s[0];
#pragma unroll
        for (int o = 1; o < 6; ++o) mx = fmaxf(mx, s[o]);
        float ex[6], sum = 0.f;
#pragma unroll
        for (int o = 0; o < 6; ++o) { ex[o] = __expf(s[o] - mx); sum += ex[o]; }
        float inv = 1.f / sum;
        if (lane == 0) {
#pragma unroll
            for (int o = 0; o < 6; ++o) sg[wave][d * 6 + o] = ex[o] * inv;
        }
    }
    // local gate (8-way over domains)
    {
        float l0 = row[512 + lane], l1 = row[544 + lane];
        float t[8];
#pragma unroll
        for (int d = 0; d < 8; ++d)
            t[d] = l0 * lW2[lane * 8 + d] + l1 * lW2[(lane + 32) * 8 + d];
#pragma unroll
        for (int m = 16; m >= 1; m >>= 1)
#pragma unroll
            for (int d = 0; d < 8; ++d) t[d] += __shfl_xor(t[d], m, 32);
#pragma unroll
        for (int d = 0; d < 8; ++d) t[d] += lb2[d];
        float mx = t[0];
#pragma unroll
        for (int d = 1; d < 8; ++d) mx = fmaxf(mx, t[d]);
        float ex[8], sum = 0.f;
#pragma unroll
        for (int d = 0; d < 8; ++d) { ex[d] = __expf(t[d] - mx); sum += ex[d]; }
        float inv = 1.f / sum;
        if (lane == 0) {
#pragma unroll
            for (int d = 0; d < 8; ++d) slg[wave][d] = ex[d] * inv;
        }
    }
    __syncthreads();
    // w[b, e']: e'<4 shared experts, else domain expert (d, e2)
    if (lane < 20) {
        float val;
        if (lane < 4) {
            val = 0.f;
#pragma unroll
            for (int d = 0; d < 8; ++d) val += slg[wave][d] * sg[wave][d * 6 + lane];
        } else {
            int idx = lane - 4, d = idx >> 1, e2 = idx & 1;
            val = slg[wave][d] * sg[wave][d * 6 + 4 + e2];
        }
        wws[(size_t)b * 32 + lane] = val;
    }
}

// ---------------------------------------------------------------------------
// Expert kernel: block = (expert e, 64-row block mb). Fuses:
//   h1 = relu(x @ eW1[e] + eb1[e])  (64x512, kept bf16 in LDS as A-fragments)
//   h2 = relu(h1 @ eW2[e] + eb2[e]) (64x256, kept in accumulators)
//   q[b,e] = h2[b,:] . tW[dom[b],:] (LDS float-atomic reduction)
// x tile staged by the Tensor Data Mover (TENSORcnt path).
// Dynamic LDS: 128K (x tiles) + 64K (h1 tiles) + 256 (q) + 256 (dom) = 197120 B
// ---------------------------------------------------------------------------
__global__ __launch_bounds__(256) void expert_kernel(const unsigned short* __restrict__ xp,
                                                     const unsigned short* __restrict__ w1p,
                                                     const unsigned short* __restrict__ w2p,
                                                     const float* __restrict__ eb1,
                                                     const float* __restrict__ eb2,
                                                     const float* __restrict__ tW,
                                                     const int* __restrict__ dom,
                                                     float* __restrict__ qws) {
    extern __shared__ char smem[];
    unsigned short* sx  = (unsigned short*)smem;               // 131072 B
    unsigned short* sh1 = (unsigned short*)(smem + 131072);    // 65536 B
    float* sq  = (float*)(smem + 196608);                      // 64 floats
    int*   sdm = (int*)  (smem + 196864);                      // 64 ints

    int e = blockIdx.x, mb = blockIdx.y;
    int tid = threadIdx.x, lane = tid & 31, wave = tid >> 5;
    int l15 = lane & 15, lh = lane >> 4;

    // ---- stage x tiles via TDM (wave 0), zero q partials, stage domain ids
    if (wave == 0) {
        tdm_stage_128k(sx, xp + (size_t)mb * 65536);
    }
    if (tid < 64) { sq[tid] = 0.f; sdm[tid] = dom[mb * 64 + tid]; }
    __syncthreads();

    // ---- phase 1: h1 = relu(x @ W1 + b1).  Wave owns 4 n-tiles, all 4 m-tiles.
    v8f acc[4][4];
#pragma unroll
    for (int m = 0; m < 4; ++m)
#pragma unroll
        for (int i = 0; i < 4; ++i) acc[m][i] = (v8f){0,0,0,0,0,0,0,0};

    const unsigned short* w1base = w1p + (size_t)e * 524288;   // 1 MB / expert
    int nbase = wave * 4;
    for (int kt = 0; kt < 32; ++kt) {
        v16bf bfrag[4], afrag[4];
#pragma unroll
        for (int i = 0; i < 4; ++i)
            bfrag[i] = *(const v16bf*)(w1base + (size_t)((nbase + i) * 32 + kt) * 512 + lane * 16);
#pragma unroll
        for (int m = 0; m < 4; ++m)
            afrag[m] = *(const v16bf*)(sx + (size_t)(m * 32 + kt) * 512 + lane * 16);
#pragma unroll
        for (int m = 0; m < 4; ++m)
#pragma unroll
            for (int i = 0; i < 4; ++i)
                acc[m][i] = wmma_bf16(afrag[m], bfrag[i], acc[m][i]);
    }

    // epilogue: relu+bias, scatter into A-fragment layout of h1 in LDS
#pragma unroll
    for (int i = 0; i < 4; ++i) {
        int n = (nbase + i) * 16 + l15;           // h1 column = K of phase-2 GEMM
        float bias = eb1[e * H1_ + n];
        int kt2 = n >> 5, kk = n & 31;
        int j = ((kk & 16) ? 8 : 0) + (kk & 7);
        int khibit = (kk >> 3) & 1;
#pragma unroll
        for (int m = 0; m < 4; ++m)
#pragma unroll
            for (int r = 0; r < 8; ++r) {
                float v = fmaxf(acc[m][i][r] + bias, 0.f);
                int mrow = r + 8 * lh;
                sh1[(size_t)(m * 16 + kt2) * 512 + (mrow + 16 * khibit) * 16 + j] = f2bf(v);
            }
    }
    __syncthreads();

    // ---- phase 2: h2 = relu(h1 @ W2 + b2).  Wave owns 2 n-tiles, all 4 m-tiles.
    v8f acc2[4][2];
#pragma unroll
    for (int m = 0; m < 4; ++m)
#pragma unroll
        for (int i = 0; i < 2; ++i) acc2[m][i] = (v8f){0,0,0,0,0,0,0,0};

    const unsigned short* w2base = w2p + (size_t)e * 131072;   // 256 KB / expert
    int n2base = wave * 2;
    for (int kt = 0; kt < 16; ++kt) {
        v16bf b2f[2], a2f[4];
#pragma unroll
        for (int i = 0; i < 2; ++i)
            b2f[i] = *(const v16bf*)(w2base + (size_t)((n2base + i) * 16 + kt) * 512 + lane * 16);
#pragma unroll
        for (int m = 0; m < 4; ++m)
            a2f[m] = *(const v16bf*)(sh1 + (size_t)(m * 16 + kt) * 512 + lane * 16);
#pragma unroll
        for (int m = 0; m < 4; ++m)
#pragma unroll
            for (int i = 0; i < 2; ++i)
                acc2[m][i] = wmma_bf16(a2f[m], b2f[i], acc2[m][i]);
    }

    // ---- phase 3: q partials = relu(h2) . tW[dom[b]]
    int   ncol2[2];
    float bias2[2];
#pragma unroll
    for (int i = 0; i < 2; ++i) {
        ncol2[i] = (n2base + i) * 16 + l15;
        bias2[i] = eb2[e * H2_ + ncol2[i]];
    }
#pragma unroll
    for (int m = 0; m < 4; ++m) {
#pragma unroll
        for (int r = 0; r < 8; ++r) {
            int mrow = r + 8 * lh;
            int bloc = m * 16 + mrow;
            int d = sdm[bloc];
            float v0 = fmaxf(acc2[m][0][r] + bias2[0], 0.f) * tW[d * H2_ + ncol2[0]];
            float v1 = fmaxf(acc2[m][1][r] + bias2[1], 0.f) * tW[d * H2_ + ncol2[1]];
            atomicAdd(&sq[bloc], v0 + v1);
        }
    }
    __syncthreads();
    if (tid < 64) qws[(size_t)(mb * 64 + tid) * 32 + e] = sq[tid];
}

// ---------------------------------------------------------------------------
// Final: out[b] = sigmoid( sum_e w[b,e]*q[b,e] + tb[dom[b]] )
// ---------------------------------------------------------------------------
__global__ __launch_bounds__(256) void final_kernel(const float* __restrict__ wws,
                                                    const float* __restrict__ qws,
                                                    const float* __restrict__ tb,
                                                    const int* __restrict__ dom,
                                                    float* __restrict__ out) {
    int b = blockIdx.x * 256 + threadIdx.x;
    if (b >= BB) return;
    float s = 0.f;
#pragma unroll
    for (int e = 0; e < NE_; ++e)
        s += wws[(size_t)b * 32 + e] * qws[(size_t)b * 32 + e];
    s += tb[dom[b]];
    out[b] = 1.f / (1.f + __expf(-s));
}

// ---------------------------------------------------------------------------
// Host launcher
// ---------------------------------------------------------------------------
extern "C" void kernel_launch(void* const* d_in, const int* in_sizes, int n_in,
                              void* d_out, int out_size, void* d_ws, size_t ws_size,
                              hipStream_t stream) {
    (void)in_sizes; (void)n_in; (void)out_size; (void)ws_size;
    const float* x    = (const float*)d_in[0];
    const int*   dom  = (const int*)  d_in[1];
    const float* eW1  = (const float*)d_in[2];
    const float* eb1  = (const float*)d_in[3];
    const float* eW2  = (const float*)d_in[4];
    const float* eb2  = (const float*)d_in[5];
    const float* gW1  = (const float*)d_in[6];
    const float* gb1  = (const float*)d_in[7];
    const float* gW2  = (const float*)d_in[8];
    const float* gb2  = (const float*)d_in[9];
    const float* lW1  = (const float*)d_in[10];
    const float* lb1  = (const float*)d_in[11];
    const float* lW2  = (const float*)d_in[12];
    const float* lb2  = (const float*)d_in[13];
    const float* tW   = (const float*)d_in[14];
    const float* tb   = (const float*)d_in[15];
    float* out = (float*)d_out;

    // workspace layout (bytes)
    char* ws = (char*)d_ws;
    unsigned short* xp  = (unsigned short*)(ws);                 // 16 MB
    unsigned short* w1p = (unsigned short*)(ws + 16777216);      // 20 MB
    unsigned short* w2p = (unsigned short*)(ws + 37748736);      // 5 MB
    unsigned short* gwp = (unsigned short*)(ws + 42991616);      // 1.125 MB
    float* ghws = (float*)(ws + 44171264);                       // 18 MB
    float* qws  = (float*)(ws + 63045632);                       // 1 MB
    float* wws  = (float*)(ws + 64094208);                       // 1 MB

    // ---- packing (bf16, fragment-linear)
    pack_x_kernel<<<2048, 256, 0, stream>>>(x, xp);
    // eW1: 20 batches, K=1024 N=512 -> nTilesK=32, tiles/batch=32*32=1024
    pack_B_kernel<<<2560, 256, 0, stream>>>(eW1, w1p, 512, 32, (long)IN_ * H1_, 1024, 20480);
    // eW2: 20 batches, K=512 N=256 -> nTilesK=16, tiles/batch=16*16=256
    pack_B_kernel<<<640, 256, 0, stream>>>(eW2, w2p, 256, 16, (long)H1_ * H2_, 256, 5120);
    // gW1: 8 batches, K=1024 N=64 -> nTilesK=32, tiles/batch=4*32=128
    pack_B_kernel<<<128, 256, 0, stream>>>(gW1, gwp, 64, 32, (long)IN_ * GG, 128, 1024);
    // lW1: 1 batch appended after the 8 domain blocks (8*128 tiles * 512 ushorts)
    pack_B_kernel<<<16, 256, 0, stream>>>(lW1, gwp + (size_t)8 * 128 * 512, 64, 32,
                                          (long)IN_ * GG, 128, 128);

    // ---- gate path
    gate_gemm_kernel<<<128, 256, 131072, stream>>>(xp, gwp, gb1, lb1, ghws);
    gate_finish_kernel<<<1024, 256, 0, stream>>>(ghws, gW2, gb2, lW2, lb2, wws);

    // ---- expert path (dominant WMMA work)
    expert_kernel<<<dim3(20, 128), 256, 197120, stream>>>(xp, w1p, w2p, eb1, eb2, tW, dom, qws);

    // ---- combine
    final_kernel<<<32, 256, 0, stream>>>(wws, qws, tb, dom, out);
}
